// MyLoss_45518063403453
// MI455X (gfx1250) — compile-verified, matrix operand order checked by reading, not run on hardware
//
#include <hip/hip_runtime.h>

#define MARGIN 0.1f
#define G 128

typedef __attribute__((ext_vector_type(2)))  float    v2f;
typedef __attribute__((ext_vector_type(8)))  float    v8f;
typedef __attribute__((ext_vector_type(16))) _Float16 v16h;

// ---------------------------------------------------------------------------
// ws[0] accumulator init (graph-capture safe: plain kernel, no memset API)
// ---------------------------------------------------------------------------
__global__ void mrl_init_ws(double* ws) {
    if (threadIdx.x == 0 && blockIdx.x == 0) ws[0] = 0.0;
}

// ---------------------------------------------------------------------------
// Main kernel: one wave32 owns a 16-row tile of the [N, 128] score matrix.
// hinge(i,j) = max(0, MARGIN - correct_i + x[i,j]) computed in f32,
// row-reduced on the matrix pipe via WMMA with an all-ones B matrix.
// Sum of all 256 f32 accumulator slots == 16 * tile_sum.
// ---------------------------------------------------------------------------
__global__ __launch_bounds__(256) void mrl_main(const float* __restrict__ inputs,
                                                const int*   __restrict__ labels,
                                                double*      __restrict__ ws,
                                                int nrows) {
    const int lane            = threadIdx.x & 31;
    const int wave_in_block   = threadIdx.x >> 5;
    const int waves_per_block = blockDim.x >> 5;
    const int gwave           = blockIdx.x * waves_per_block + wave_in_block;
    const int nwaves          = gridDim.x * waves_per_block;

    const int r    = lane & 15;   // row within the 16-row tile
    const int half = lane >> 4;   // half-wave: selects K slice of A
    const int ntiles = nrows >> 4;

    for (int t = gwave; t < ntiles; t += nwaves) {   // wave-uniform -> EXEC all-1s
        const int   row     = (t << 4) + r;
        const int   lab     = labels[row];
        const float correct = inputs[(size_t)row * G + lab];
        const float bias    = MARGIN - correct;
        const float* rp     = inputs + (size_t)row * G;

        v8f acc = {};

#if __has_builtin(__builtin_amdgcn_wmma_f32_16x16x4_f32)
        // f32 path: A = 16x4 hinge chunk (2 VGPRs), B = 4x16 ones (2 VGPRs).
        v2f onesv = {1.0f, 1.0f};
        #pragma unroll
        for (int k = 0; k < 32; ++k) {
            const int col = (k << 2) + (half << 1);        // lanes 0-15: K=0,1 ; 16-31: K=2,3
            v2f x = *(const v2f*)(rp + col);               // global_load_b64
            v2f a;
            a[0] = fmaxf(x[0] + bias, 0.0f);
            a[1] = fmaxf(x[1] + bias, 0.0f);
            acc = __builtin_amdgcn_wmma_f32_16x16x4_f32(
                /*neg_a=*/false, a, /*neg_b=*/false, onesv,
                /*c_mod=*/(short)0, acc, /*reuse_a=*/false, /*reuse_b=*/false);
        }
#else
        // Fallback: codegen-confirmed f16 WMMA (K=32). Hinge in f32, cast to
        // f16 only for the A operand; accumulation stays f32.
        v16h onesh;
        #pragma unroll
        for (int i = 0; i < 16; ++i) onesh[i] = (_Float16)1.0f;
        #pragma unroll
        for (int kk = 0; kk < 4; ++kk) {
            // lane r   (half=0): K = 0..7  and 16..23 of this 32-chunk
            // lane r+16(half=1): K = 8..15 and 24..31
            const int c0 = (kk << 5) + (half << 3);
            v16h a;
            #pragma unroll
            for (int j = 0; j < 8; ++j) {
                a[j]     = (_Float16)fmaxf(rp[c0 + j]      + bias, 0.0f);
                a[8 + j] = (_Float16)fmaxf(rp[c0 + 16 + j] + bias, 0.0f);
            }
            acc = __builtin_amdgcn_wmma_f32_16x16x32_f16(
                false, a, false, onesh, (short)0, acc, false, false);
        }
#endif

        // Every D column holds the row sum -> total over all 256 slots = 16*tile_sum.
        float s = acc[0] + acc[1] + acc[2] + acc[3]
                + acc[4] + acc[5] + acc[6] + acc[7];
        #pragma unroll
        for (int off = 16; off > 0; off >>= 1)
            s += __shfl_down(s, off, 32);

        if (lane == 0)
            atomicAdd(ws, (double)(s * (1.0f / 16.0f)));   // global_atomic_add_f64
    }
}

// ---------------------------------------------------------------------------
// Finalize: remove the j==label contribution (exactly MARGIN per row) and
// divide by the pair count.
// ---------------------------------------------------------------------------
__global__ void mrl_finalize(const double* __restrict__ ws,
                             float* __restrict__ out, int n) {
    if (threadIdx.x == 0 && blockIdx.x == 0) {
        double total = ws[0] - (double)n * (double)MARGIN;
        out[0] = (float)(total / ((double)n * (double)(G - 1)));
    }
}

// ---------------------------------------------------------------------------
extern "C" void kernel_launch(void* const* d_in, const int* in_sizes, int n_in,
                              void* d_out, int out_size, void* d_ws, size_t ws_size,
                              hipStream_t stream) {
    const float* inputs = (const float*)d_in[0];
    const int*   labels = (const int*)d_in[1];
    float*       out    = (float*)d_out;
    double*      acc    = (double*)d_ws;

    const int n = in_sizes[1];            // N = 262144 rows; in_sizes[0] = N*G

    mrl_init_ws<<<1, 1, 0, stream>>>(acc);

    // 8 waves/block; 1024 blocks -> 8192 waves, grid-stride over N/16 tiles.
    const int block  = 256;
    int       blocks = 1024;
    const int ntiles = n >> 4;
    const int wavecap = (ntiles + 7) / 8;     // don't over-launch tiny problems
    if (blocks > wavecap) blocks = (wavecap > 0) ? wavecap : 1;

    mrl_main<<<blocks, block, 0, stream>>>(inputs, labels, acc, n);
    mrl_finalize<<<1, 1, 0, stream>>>(acc, out, n);
}
